// MonitoredSelfAttention_38920993636369
// MI455X (gfx1250) — matrix operand order repaired and checked by reading, not compiled, not run on hardware
//
#include <hip/hip_runtime.h>
#include <hip/hip_bf16.h>

typedef __bf16 bf16_t;
typedef __attribute__((ext_vector_type(16))) __bf16 v16bf;
typedef __attribute__((ext_vector_type(8)))  float  v8f;

#define BATCH 2
#define SEQ   2048
#define CH    1024
#define NHEAD 16
#define HDIM  64

// ---------------------------------------------------------------------------
// WMMA fragment helpers (gfx1250 wave32, 16x16x32 bf16 -> f32). K order is
// this code's contiguous K=16h+e (not ISA table order): A 16x32 at M=lane&15,
// B 32x16 with K=16h+e at N=lane&15, C/D row m=r+8*(lane>>4), col n=lane&15.
// ---------------------------------------------------------------------------
static __device__ __forceinline__ v16bf wmma_frag_a(const bf16_t* p0, int ld, int lane) {
  int h = lane >> 4, m = lane & 15;
  const bf16_t* p = p0 + m * ld + 16 * h;
  v16bf a;
#pragma unroll
  for (int e = 0; e < 16; ++e) a[e] = p[e];
  return a;
}

// B fragment from [n][k] storage (math-B transposed, contiguous in k)
static __device__ __forceinline__ v16bf wmma_frag_b_nk(const bf16_t* p0, int ld, int lane) {
  int h = lane >> 4, n = lane & 15;
  const bf16_t* p = p0 + n * ld + 16 * h;
  v16bf b;
#pragma unroll
  for (int e = 0; e < 16; ++e) b[e] = p[e];
  return b;
}

// B fragment from [k][n] storage (math-B row-major)
static __device__ __forceinline__ v16bf wmma_frag_b_kn(const bf16_t* p0, int ld, int lane) {
  int h = lane >> 4, n = lane & 15;
  const bf16_t* p = p0 + (16 * h) * ld + n;
  v16bf b;
#pragma unroll
  for (int e = 0; e < 16; ++e) b[e] = p[e * ld];
  return b;
}

static __device__ __forceinline__ v8f wmma_bf16(v16bf a, v16bf b, v8f c) {
  return __builtin_amdgcn_wmma_f32_16x16x32_bf16(false, a, false, b, (short)0, c, false, false);
}

// ---------------------------------------------------------------------------
// DPP16 ROW_XMASK xor-shuffle (pure VALU, no LDS port / DScnt stalls).
// xor mask <= 15 stays inside the 16-lane DPP row, which matches the 16-lane
// groups holding one matrix row in the WMMA C layout.
// ---------------------------------------------------------------------------
template <int MASK>
static __device__ __forceinline__ float dpp_xor_f(float v) {
  int i = __builtin_bit_cast(int, v);
  int r = __builtin_amdgcn_update_dpp(0, i, 0x160 | MASK, 0xf, 0xf, true);
  return __builtin_bit_cast(float, r);
}
static __device__ __forceinline__ float reduce_max16(float v) {
  v = fmaxf(v, dpp_xor_f<1>(v));
  v = fmaxf(v, dpp_xor_f<2>(v));
  v = fmaxf(v, dpp_xor_f<4>(v));
  v = fmaxf(v, dpp_xor_f<8>(v));
  return v;
}
static __device__ __forceinline__ float reduce_add16(float v) {
  v += dpp_xor_f<1>(v);
  v += dpp_xor_f<2>(v);
  v += dpp_xor_f<4>(v);
  v += dpp_xor_f<8>(v);
  return v;
}

// ---------------------------------------------------------------------------
// Tensor Data Mover: 2D tile load (global -> LDS) via D# descriptor
// (ISA 08_async_tensor 8.3/8.4). data_size=2B. Row padding via pad_interval /
// pad_amount (DWORD units). Issued by one wave; TENSORcnt completes in-order,
// enabling ping-pong double buffering with s_wait_tensorcnt(2).
// 6-arg builtin: (u32x4 g0, i32x8 g1, i32x4, i32x4, i32x8, i32 cpol)
// ---------------------------------------------------------------------------
#if defined(__has_builtin)
#if __has_builtin(__builtin_amdgcn_tensor_load_to_lds)
#define HAVE_TDM 1
#endif
#endif

#ifdef HAVE_TDM
typedef __attribute__((ext_vector_type(4))) unsigned int tdm_g0_t;
typedef __attribute__((ext_vector_type(8))) int          tdm_g1_t;
typedef __attribute__((ext_vector_type(4))) int          tdm_g2_t;
typedef __attribute__((ext_vector_type(8))) int          tdm_g4_t;

static __device__ __forceinline__ void tdm_load_2d(
    unsigned lds_byte_off, const void* gptr,
    unsigned tensor_d0, unsigned tensor_d1, unsigned long long stride0,
    unsigned tile_d0, unsigned tile_d1,
    unsigned pad_interval_code, unsigned pad_amount_code) {
  unsigned long long ga = (unsigned long long)(uintptr_t)gptr;
  tdm_g0_t g0;
  g0[0] = 1u;                                        // count=1, user descriptor
  g0[1] = lds_byte_off;                              // lds_addr
  g0[2] = (unsigned)(ga & 0xFFFFFFFFu);              // global_addr lo
  g0[3] = (unsigned)((ga >> 32) & 0x01FFFFFFu) | 0x80000000u;  // addr hi | type=2
  unsigned flags = (1u << 16)                        // data_size: 2 bytes
                 | (1u << 20)                        // pad_enable
                 | (pad_interval_code << 22)
                 | (pad_amount_code << 25);
  tdm_g1_t g1;
  g1[0] = (int)flags;
  g1[1] = (int)((tensor_d0 & 0xFFFFu) << 16);                         // tdim0 lo
  g1[2] = (int)((tensor_d0 >> 16) | ((tensor_d1 & 0xFFFFu) << 16));   // tdim0 hi | tdim1 lo
  g1[3] = (int)((tensor_d1 >> 16) | (tile_d0 << 16));                 // tdim1 hi | tile0
  g1[4] = (int)(tile_d1 & 0xFFFFu);                                   // tile1 (tile2=0)
  g1[5] = (int)(unsigned)(stride0 & 0xFFFFFFFFull);                   // dim0 stride lo
  g1[6] = (int)(unsigned)((stride0 >> 32) & 0xFFFFull);               // stride hi
  g1[7] = 0;
  tdm_g2_t gz = {0, 0, 0, 0};
  tdm_g4_t gz8 = {0, 0, 0, 0, 0, 0, 0, 0};
  __builtin_amdgcn_tensor_load_to_lds(g0, g1, gz, gz, gz8, 0);
}
#endif

// ---------------------------------------------------------------------------
// fp32 -> bf16 streaming cast + scalar zero
// ---------------------------------------------------------------------------
__global__ void cast_f32_bf16(const float* __restrict__ src, bf16_t* __restrict__ dst, int n) {
  int i = blockIdx.x * blockDim.x + threadIdx.x;
  int stride = gridDim.x * blockDim.x;
  for (; i < n; i += stride) dst[i] = (bf16_t)src[i];
}

__global__ void zero_f32(float* p) { p[0] = 0.f; }

// ---------------------------------------------------------------------------
// Tiled bf16 GEMM, fp32 accumulate. 256 thr = 8 waves, 128x128 tile, K-step 32.
// Double-buffered LDS (dynamic, base 0), per buffer:
//   sA [128][40] bf16 (10240 B) + sB [32][136] bf16 (8704 B) = 18944 B
// Total 37888 B. TDM pre-issues the next K tile while the current is consumed.
// MODE 0: fp32 row-major out.  MODE 1: QKV scatter (block-uniform q/k/v select).
// ---------------------------------------------------------------------------
#define GA_LD 40          // 32 + 8 pad
#define GB_LD 136         // 128 + 8 pad
#define G_BUF_BYTES 18944
#define GSB_OFF 10240
#define GEMM_LDS_BYTES (2 * G_BUF_BYTES)

template <int MODE>
__global__ __launch_bounds__(256) void gemm_bf16(
    const bf16_t* __restrict__ A, const bf16_t* __restrict__ Bm,
    int N, int K,
    float* __restrict__ outF,
    bf16_t* __restrict__ qd, bf16_t* __restrict__ kd, bf16_t* __restrict__ vd) {
  extern __shared__ char smem[];

  int tid = threadIdx.x;
  int lane = tid & 31, w = tid >> 5;
  int wr = w & 3, wc = w >> 2;
  int row0 = blockIdx.y * 128;
  int col0 = blockIdx.x * 128;

  v8f acc[2][4] = {};

#ifdef HAVE_TDM
  auto issue_tiles = [&](int buf, int kt) {
    // A tile: 32 (k) x 128 (rows); LDS rows 16 DW + 4 DW pad -> codes 3,3
    tdm_load_2d(buf * G_BUF_BYTES, A + (size_t)row0 * K + kt, K, 128,
                (unsigned long long)K, 32, 128, 3, 3);
    // B tile: 128 (n) x 32 (k); LDS rows 64 DW + 4 DW pad -> codes 5,3
    tdm_load_2d(buf * G_BUF_BYTES + GSB_OFF, Bm + (size_t)kt * N + col0, N, 32,
                (unsigned long long)N, 128, 32, 5, 3);
  };
  if (w == 0) issue_tiles(0, 0);
#endif

  int cur = 0;
  for (int kt = 0; kt < K; kt += 32) {
    bf16_t* sA = (bf16_t*)(smem + cur * G_BUF_BYTES);
    bf16_t* sB = (bf16_t*)(smem + cur * G_BUF_BYTES + GSB_OFF);
#ifdef HAVE_TDM
    if (w == 0) {
      if (kt + 32 < K) {
        issue_tiles(cur ^ 1, kt + 32);             // prefetch next tile pair
        __builtin_amdgcn_s_wait_tensorcnt(2);      // oldest 2 (= current) done
      } else {
        __builtin_amdgcn_s_wait_tensorcnt(0);
      }
    }
    if (w == 7 && kt + 32 < K) {                   // warm L2 for the next tile (kt+32)
      __builtin_prefetch(A + (size_t)(row0 + lane * 4) * K + kt + 32, 0, 1);
      __builtin_prefetch(Bm + (size_t)(kt + 32 + lane) * N + col0, 0, 1);
    }
#else
    {
      int r = tid >> 1, ks = (tid & 1) * 16;
      const bf16_t* g = A + (size_t)(row0 + r) * K + kt + ks;
#pragma unroll
      for (int e = 0; e < 16; ++e) sA[r * GA_LD + ks + e] = g[e];
      int kk = tid >> 3, ns = (tid & 7) * 16;
      const bf16_t* gb = Bm + (size_t)(kt + kk) * N + col0 + ns;
#pragma unroll
      for (int e = 0; e < 16; ++e) sB[kk * GB_LD + ns + e] = gb[e];
    }
#endif
    __syncthreads();

    v16bf af[2], bfr[4];
#pragma unroll
    for (int i = 0; i < 2; ++i)
      af[i] = wmma_frag_a(sA + (wr * 32 + i * 16) * GA_LD, GA_LD, lane);
#pragma unroll
    for (int j = 0; j < 4; ++j)
      bfr[j] = wmma_frag_b_kn(sB + wc * 64 + j * 16, GB_LD, lane);
#pragma unroll
    for (int i = 0; i < 2; ++i)
#pragma unroll
      for (int j = 0; j < 4; ++j)
        acc[i][j] = wmma_bf16(af[i], bfr[j], acc[i][j]);
    __syncthreads();
    cur ^= 1;
  }

  int h = lane >> 4, n = lane & 15;
  if (MODE == 0) {
#pragma unroll
    for (int i = 0; i < 2; ++i)
#pragma unroll
      for (int j = 0; j < 4; ++j)
#pragma unroll
        for (int r = 0; r < 8; ++r) {
          int rr = row0 + wr * 32 + i * 16 + r + 8 * h;
          int cc = col0 + wc * 64 + j * 16 + n;
          outF[(size_t)rr * N + cc] = acc[i][j][r];
        }
  } else {
    // Block-uniform destination: the 128-col tile sits inside exactly one of
    // q/k/v (128 | 1024) -> no per-element branching.
    int which = col0 >> 10;
    bf16_t* dst = (which == 0) ? qd : (which == 1) ? kd : vd;
    int cbase = col0 & (CH - 1);
#pragma unroll
    for (int i = 0; i < 2; ++i)
#pragma unroll
      for (int j = 0; j < 4; ++j)
#pragma unroll
        for (int r = 0; r < 8; ++r) {
          int rr = row0 + wr * 32 + i * 16 + r + 8 * h;
          int c2 = cbase + wc * 64 + j * 16 + n;
          int t = rr & (SEQ - 1);
          int bb = rr >> 11;
          int hh = c2 >> 6, dd = c2 & 63;
          size_t di = (((size_t)bb * NHEAD + hh) * SEQ + t) * HDIM + dd;
          dst[di] = (bf16_t)acc[i][j][r];
        }
  }
}

// ---------------------------------------------------------------------------
// Fused causal flash attention + entropy.
// Grid (T/64, H, B); 128 thr = 4 waves, 16 query rows each.
// Double-buffered K/V staging via TDM; per buffer K[32][72] + V[32][72] bf16
// (9216 B). sP 4x[16][40] at 18432. Total 23552 B.
// ---------------------------------------------------------------------------
#define FA_LDKV 72
#define FA_LDP  40
#define F_KV_BYTES 4608
#define F_BUF_BYTES 9216
#define FSP_OFF 18432
#define FA_LDS_BYTES (FSP_OFF + 4 * 16 * FA_LDP * 2)

__global__ __launch_bounds__(128) void flash_attn(
    const bf16_t* __restrict__ Q, const bf16_t* __restrict__ Kt,
    const bf16_t* __restrict__ Vt, bf16_t* __restrict__ O,
    float* __restrict__ entAcc) {
  extern __shared__ char smem[];

  int tid = threadIdx.x;
  int lane = tid & 31, w = tid >> 5;
  int h = lane >> 4, n = lane & 15;
  bf16_t* sP = (bf16_t*)(smem + FSP_OFF) + w * (16 * FA_LDP);
  int qt = blockIdx.x, head = blockIdx.y, bb = blockIdx.z;

  size_t base = ((size_t)bb * NHEAD + head) * SEQ * HDIM;
  const bf16_t* Qh = Q + base;
  const bf16_t* Kh = Kt + base;
  const bf16_t* Vh = Vt + base;
  int qrow0 = qt * 64 + w * 16;
  int kend = qt * 64 + 64;

  v16bf aq[2];
  aq[0] = wmma_frag_a(Qh + (size_t)qrow0 * HDIM + 0, HDIM, lane);
  aq[1] = wmma_frag_a(Qh + (size_t)qrow0 * HDIM + 32, HDIM, lane);

  float mrow[8], lrow[8], wrow[8];
#pragma unroll
  for (int r = 0; r < 8; ++r) { mrow[r] = -3.0e38f; lrow[r] = 0.f; wrow[r] = 0.f; }
  v8f o[4] = {};

#ifdef HAVE_TDM
  auto issue_kv = [&](int buf, int ks) {
    // 64 (d) x 32 (keys) tiles; LDS rows 32 DW + 4 DW pad -> codes 4,3
    tdm_load_2d(buf * F_BUF_BYTES, Kh + (size_t)ks * HDIM,
                HDIM, SEQ, HDIM, HDIM, 32, 4, 3);
    tdm_load_2d(buf * F_BUF_BYTES + F_KV_BYTES, Vh + (size_t)ks * HDIM,
                HDIM, SEQ, HDIM, HDIM, 32, 4, 3);
  };
  if (w == 0) issue_kv(0, 0);
#endif

  int cur = 0;
  for (int ks = 0; ks < kend; ks += 32) {
    bf16_t* sK = (bf16_t*)(smem + cur * F_BUF_BYTES);
    bf16_t* sV = (bf16_t*)(smem + cur * F_BUF_BYTES + F_KV_BYTES);
#ifdef HAVE_TDM
    if (w == 0) {
      if (ks + 32 < kend) {
        issue_kv(cur ^ 1, ks + 32);
        __builtin_amdgcn_s_wait_tensorcnt(2);
      } else {
        __builtin_amdgcn_s_wait_tensorcnt(0);
      }
    }
#else
    {
      int kk = tid >> 2, ds = (tid & 3) * 16;
      const bf16_t* gk = Kh + (size_t)(ks + kk) * HDIM + ds;
      const bf16_t* gv = Vh + (size_t)(ks + kk) * HDIM + ds;
#pragma unroll
      for (int e = 0; e < 16; ++e) {
        sK[kk * FA_LDKV + ds + e] = gk[e];
        sV[kk * FA_LDKV + ds + e] = gv[e];
      }
    }
#endif
    __syncthreads();

#pragma unroll
    for (int sub = 0; sub < 2; ++sub) {
      v8f s = {};
      // S = Q * K^T ; sK row-major [key][d] == B^T [n][k]
      s = wmma_bf16(aq[0], wmma_frag_b_nk(sK + (sub * 16) * FA_LDKV + 0, FA_LDKV, lane), s);
      s = wmma_bf16(aq[1], wmma_frag_b_nk(sK + (sub * 16) * FA_LDKV + 32, FA_LDKV, lane), s);
      int kb = ks + sub * 16;
#pragma unroll
      for (int r = 0; r < 8; ++r) {
        float sv = s[r] * 0.125f;             // 1/sqrt(64)
        int qi = qrow0 + r + 8 * h;
        int kj = kb + n;
        if (kj > qi) sv = -3.0e38f;           // finite mask: exp->0, no NaN
        float mx = reduce_max16(sv);          // DPP ROW_XMASK, no LDS traffic
        float mnew = fmaxf(mrow[r], mx);
        float corr = __expf(mrow[r] - mnew);
        lrow[r] *= corr;
        wrow[r] *= corr;
#pragma unroll
        for (int j = 0; j < 4; ++j) o[j][r] *= corr;
        mrow[r] = mnew;
        float p = __expf(sv - mnew);
        lrow[r] += reduce_add16(p);
        wrow[r] += reduce_add16(p * sv);
        sP[(r + 8 * h) * FA_LDP + sub * 16 + n] = (bf16_t)p;
      }
    }

    // O += P(16x32) * V(32x64); sV row-major [key][d] == B [k][n]
    v16bf pa = wmma_frag_a(sP, FA_LDP, lane);
#pragma unroll
    for (int j = 0; j < 4; ++j)
      o[j] = wmma_bf16(pa, wmma_frag_b_kn(sV + j * 16, FA_LDKV, lane), o[j]);
    __syncthreads();
    cur ^= 1;
  }

#pragma unroll
  for (int r = 0; r < 8; ++r) {
    float inv = 1.f / lrow[r];
    if (n == 0) {  // lane 0: rows 0-7, lane 16: rows 8-15
      float Hrow = mrow[r] + __logf(lrow[r]) - wrow[r] * inv;
      atomicAdd(entAcc, Hrow);
    }
    int grow = bb * SEQ + qt * 64 + w * 16 + r + 8 * h;
#pragma unroll
    for (int j = 0; j < 4; ++j) {
      int gcol = head * HDIM + j * 16 + n;
      O[(size_t)grow * CH + gcol] = (bf16_t)(o[j][r] * inv);
    }
  }
}

__global__ void finalize_entropy(const float* __restrict__ entAcc, float* __restrict__ out) {
  out[0] = entAcc[0] * (1.0f / (float)(BATCH * NHEAD * SEQ));
}

// ---------------------------------------------------------------------------
// Launch
// ---------------------------------------------------------------------------
extern "C" void kernel_launch(void* const* d_in, const int* in_sizes, int n_in,
                              void* d_out, int out_size, void* d_ws, size_t ws_size,
                              hipStream_t stream) {
  const float* x      = (const float*)d_in[0];
  const float* w_attn = (const float*)d_in[1];
  const float* w_proj = (const float*)d_in[2];
  float* out = (float*)d_out;

  const size_t nX  = (size_t)BATCH * SEQ * CH;
  const size_t nWa = (size_t)CH * 3 * CH;
  const size_t nWp = (size_t)CH * CH;
  const size_t nHD = (size_t)BATCH * NHEAD * SEQ * HDIM;

  char* ws = (char*)d_ws;
  size_t off = 0;
  auto salloc = [&](size_t bytes) -> void* {
    void* p = ws + off;
    off += (bytes + 255) & ~(size_t)255;
    return p;
  };
  bf16_t* xb  = (bf16_t*)salloc(nX * 2);
  bf16_t* wab = (bf16_t*)salloc(nWa * 2);
  bf16_t* wpb = (bf16_t*)salloc(nWp * 2);
  bf16_t* qb  = (bf16_t*)salloc(nHD * 2);
  bf16_t* kb  = (bf16_t*)salloc(nHD * 2);
  bf16_t* vb  = (bf16_t*)salloc(nHD * 2);
  bf16_t* ob  = (bf16_t*)salloc(nX * 2);
  float*  entAcc = (float*)salloc(256);
  (void)ws_size; (void)in_sizes; (void)n_in; (void)out_size;

  cast_f32_bf16<<<2048, 256, 0, stream>>>(x, xb, (int)nX);
  cast_f32_bf16<<<2048, 256, 0, stream>>>(w_attn, wab, (int)nWa);
  cast_f32_bf16<<<1024, 256, 0, stream>>>(w_proj, wpb, (int)nWp);
  zero_f32<<<1, 1, 0, stream>>>(entAcc);

  gemm_bf16<1><<<dim3(3 * CH / 128, BATCH * SEQ / 128), 256, GEMM_LDS_BYTES, stream>>>(
      xb, wab, 3 * CH, CH, nullptr, qb, kb, vb);

  flash_attn<<<dim3(SEQ / 64, NHEAD, BATCH), 128, FA_LDS_BYTES, stream>>>(
      qb, kb, vb, ob, entAcc);

  gemm_bf16<0><<<dim3(CH / 128, BATCH * SEQ / 128), 256, GEMM_LDS_BYTES, stream>>>(
      ob, wpb, CH, CH, out, nullptr, nullptr, nullptr);

  finalize_entropy<<<1, 1, 0, stream>>>(entAcc, out + nX);
}
